// ISTFT_4449586118853
// MI455X (gfx1250) — matrix operand order repaired and checked by reading, not compiled, not run on hardware
//
#include <hip/hip_runtime.h>
#include <hip/hip_bf16.h>

// ---------------------------------------------------------------------------
// ISTFT on MI455X (gfx1250): folded-spectrum GEMM via V_WMMA_F32_16X16X32_BF16
// with bf16 hi/lo split (3-product) for ~f32 accuracy, then overlap-add.
// Round 4: native bf16 conversions (v_cvt_pk_bf16_f32 when available) replace
// the integer-emulated rounding that dominated the VALU count.
// ---------------------------------------------------------------------------

typedef __attribute__((ext_vector_type(16))) __bf16 v16bf;
typedef __attribute__((ext_vector_type(2)))  __bf16 v2bf;
typedef __attribute__((ext_vector_type(8)))  float  v8f;

#define B_SZ    8
#define T_SZ    1024
#define NFFT    2048
#define HOP     512
#define FREQ    1025          // N_FFT/2 + 1
#define M_TOT   (B_SZ * T_SZ) // 8192
#define N_TOT   NFFT          // 2048
#define KSEG    1056          // 1025 padded to 33 k-tiles of 32
#define K_TILES 66            // real(33) + imag(33)
#define NT_TOT  (N_TOT / 16)  // 128 n-tiles
#define LEN_OUT 523776        // (T-1)*HOP
#define L_TOT   525824        // (T-1)*HOP + N_FFT

// workspace layout (bytes)
#define WHI_BYTES ((size_t)K_TILES * NT_TOT * 32 * 8 * 4)   // 8,650,752
#define WLO_OFF   WHI_BYTES
#define S_OFF     (2 * WHI_BYTES)                           // 17,301,504
// S buffer: M_TOT * N_TOT * 4 = 67,108,864  -> total ~84.4 MB

// ---- native bf16 helpers ---------------------------------------------------
__device__ __forceinline__ unsigned cvt_pack_bf16(float a, float b) {
#if __has_builtin(__builtin_amdgcn_cvt_pk_bf16_f32)
    v2bf pk = __builtin_amdgcn_cvt_pk_bf16_f32(a, b);
    return __builtin_bit_cast(unsigned, pk);
#else
    unsigned short ua = __builtin_bit_cast(unsigned short, (__bf16)a);
    unsigned short ub = __builtin_bit_cast(unsigned short, (__bf16)b);
    return (unsigned)ua | ((unsigned)ub << 16);
#endif
}
__device__ __forceinline__ float bf_hi_f32(float x) {   // f32 -> bf16 -> f32
    return (float)((__bf16)x);
}
// split x into (hi, lo) bf16 pair packed later:  x ~= hi + lo
__device__ __forceinline__ void split2(float x0, float x1,
                                       unsigned& hi, unsigned& lo) {
    float h0 = bf_hi_f32(x0), h1 = bf_hi_f32(x1);
    hi = cvt_pack_bf16(x0, x1);          // hi parts (RNE)
    lo = cvt_pack_bf16(x0 - h0, x1 - h1);
}

union FragU {
    unsigned u[8];
    uint4    q[2];
    v16bf    v;
};

// Folded weight value W[kg, n]:
//   kg <  KSEG: real part, i = kg:      Cr[n,i] + (1<=i<=1023)*Cr[n,2048-i]
//   kg >= KSEG: imag part, i = kg-KSEG: -Ci[n,i] + (1<=i<=1023)*Ci[n,2048-i]
__device__ __forceinline__ float fold_w(const float* __restrict__ Cr,
                                        const float* __restrict__ Ci,
                                        int kg, int n) {
    if (kg < KSEG) {
        int i = kg;
        if (i > 1024) return 0.f;
        float w = Cr[(size_t)n * NFFT + i];
        if (i >= 1 && i <= 1023) w += Cr[(size_t)n * NFFT + (NFFT - i)];
        return w;
    } else {
        int i = kg - KSEG;
        if (i > 1024) return 0.f;
        float w = -Ci[(size_t)n * NFFT + i];
        if (i >= 1 && i <= 1023) w += Ci[(size_t)n * NFFT + (NFFT - i)];
        return w;
    }
}

// ---------------------------------------------------------------------------
// Kernel 1: build folded weights, split into bf16 hi/lo, store in exact
// B-fragment (32x16, 16-bit) VGPR layout so GEMM loads are 2x b128 per frag.
// dword index g = ((kt*128 + nt)*32 + lane)*8 + v
// lane 0-15: cols n0..n0+15, K = 2v,2v+1 ; lane 16-31: same cols, K = 16+2v
// ---------------------------------------------------------------------------
__global__ __launch_bounds__(256) void prep_w_kernel(
    const float* __restrict__ Cr, const float* __restrict__ Ci,
    unsigned* __restrict__ Whi, unsigned* __restrict__ Wlo)
{
    int g    = blockIdx.x * 256 + threadIdx.x;
    int v    = g & 7;
    int lane = (g >> 3) & 31;
    int nt   = (g >> 8) & 127;
    int kt   = g >> 15;
    if (kt >= K_TILES) return;

    int n = nt * 16 + (lane & 15);
    int k = kt * 32 + ((lane & 16) ? 16 : 0) + 2 * v;

    float w0 = fold_w(Cr, Ci, k,     n);
    float w1 = fold_w(Cr, Ci, k + 1, n);
    unsigned hi, lo;
    split2(w0, w1, hi, lo);
    Whi[g] = hi;
    Wlo[g] = lo;
}

// ---------------------------------------------------------------------------
// Raw A fetch for one k-tile: 8 dwords worth (16 f32) per thread.
// Only kt==32 / kt==65 touch the zero-padded K region; that test is UNIFORM in
// kt (scalar branch). Hot path: 16 loads with compile-time immediate offsets.
// ---------------------------------------------------------------------------
__device__ __forceinline__ void load_a_raw(const float* __restrict__ re,
                                           const float* __restrict__ im,
                                           int kt, size_t rowoff, int kswiz,
                                           float (&x)[8][2])
{
    const float* src   = (kt <= 32) ? re : im;
    const int    ibase = ((kt <= 32) ? kt : (kt - 33)) * 32 + kswiz;

    if (kt == 32 || kt == 65) {          // uniform scalar branch: edge tiles
        const float* pb = src + rowoff;
#pragma unroll
        for (int v = 0; v < 8; ++v) {
            int koff = 2 * v;
            int i0 = ibase + koff;
            int i1 = i0 + 1;
            float x0 = pb[(i0 < 1024) ? i0 : 1024];   // clamped: always legal
            float x1 = pb[(i1 < 1024) ? i1 : 1024];
            x[v][0] = (i0 <= 1024) ? x0 : 0.f;        // v_cndmask
            x[v][1] = (i1 <= 1024) ? x1 : 0.f;
        }
    } else {                              // hot path: provably in-bounds
        const float* p = src + rowoff + ibase;
#pragma unroll
        for (int v = 0; v < 8; ++v) {
            int koff = 2 * v;
            x[v][0] = p[koff];            // immediate-offset loads
            x[v][1] = p[koff + 1];
        }
    }
}

// ---------------------------------------------------------------------------
// Kernel 2: GEMM  S[m, n] = sum_k A[m,k] * W[k,n]
// Workgroup: 256 threads (8 wave32), tile 128(M) x 128(N).
// Wave (wm 0..3, wn 0..1) computes a 32x64 sub-tile = 2x4 C frags.
// Double-buffered LDS A stage (literal stage offsets via x2 unroll);
// B frags straight from global (L2-resident).  bf16x3: C += Ahi*Bhi+Ahi*Blo+Alo*Bhi
// ---------------------------------------------------------------------------
__global__ __launch_bounds__(256) void istft_gemm_kernel(
    const float* __restrict__ re, const float* __restrict__ im,
    const unsigned* __restrict__ Whi, const unsigned* __restrict__ Wlo,
    float* __restrict__ S)
{
    __shared__ __align__(16) unsigned AshHi[4096];  // 2 stages x 2048 dwords
    __shared__ __align__(16) unsigned AshLo[4096];

    const int tid  = threadIdx.x;
    const int lane = tid & 31;
    const int wid  = tid >> 5;
    const int wm   = wid & 3;    // M sub-block (32 rows each)
    const int wn   = wid >> 2;   // N sub-block (64 cols each)
    const int bm   = blockIdx.x; // 0..63
    const int bn   = blockIdx.y; // 0..15
    const int m_base = bm * 128;

    v8f acc[2][4];
#pragma unroll
    for (int mf = 0; mf < 2; ++mf)
#pragma unroll
        for (int nf = 0; nf < 4; ++nf)
#pragma unroll
            for (int e = 0; e < 8; ++e) acc[mf][nf][e] = 0.f;

    // A-staging mapping: thread tid -> (m-tile = tid>>5, lane), 8 dwords
    const int    fmt    = tid >> 5;
    const size_t rowoff = (size_t)(m_base + fmt * 16 + (lane & 15)) * FREQ;
    const int    kswiz  = (lane & 16) ? 16 : 0;
    const int    sidx   = fmt * 256 + lane * 8;       // stage dword base
    const int    abase  = ((wm * 2) * 32 + lane) * 8; // A-frag read base (mf=0)
    const size_t woff0  = ((size_t)(bn * 8 + wn * 4) * 32 + lane) * 8; // B dword base

    float xr[8][2];
    load_a_raw(re, im, 0, rowoff, kswiz, xr);   // prefetch tile 0

    auto step = [&](int kt, int sb) {
        // ---- convert previously fetched raws, store into stage sb ----
#pragma unroll
        for (int v = 0; v < 8; ++v) {
            unsigned hi, lo;
            split2(xr[v][0], xr[v][1], hi, lo);
            AshHi[sb + sidx + v] = hi;
            AshLo[sb + sidx + v] = lo;
        }
        __syncthreads();

        // ---- issue next tile's global loads (WAR on xr, overlaps WMMA) ----
        if (kt + 1 < K_TILES)
            load_a_raw(re, im, kt + 1, rowoff, kswiz, xr);

        // ---- load B fragments straight from global (pre-laid-out) ----
        FragU bhi[4], blo[4];
        {
            const size_t kb = (size_t)kt * (NT_TOT * 32 * 8);
#pragma unroll
            for (int nf = 0; nf < 4; ++nf) {
                const uint4* p = (const uint4*)(Whi + kb + woff0 + (size_t)nf * 256);
                const uint4* q = (const uint4*)(Wlo + kb + woff0 + (size_t)nf * 256);
                bhi[nf].q[0] = p[0]; bhi[nf].q[1] = p[1];
                blo[nf].q[0] = q[0]; blo[nf].q[1] = q[1];
            }
        }

        // ---- load A fragments from stage sb (b128 DS loads) ----
        FragU ahi[2], alo[2];
#pragma unroll
        for (int mf = 0; mf < 2; ++mf) {
            const uint4* ph = (const uint4*)&AshHi[sb + abase + mf * 256];
            const uint4* pl = (const uint4*)&AshLo[sb + abase + mf * 256];
            ahi[mf].q[0] = ph[0]; ahi[mf].q[1] = ph[1];
            alo[mf].q[0] = pl[0]; alo[mf].q[1] = pl[1];
        }

        // ---- WMMA: bf16x3 ----
#pragma unroll
        for (int mf = 0; mf < 2; ++mf) {
#pragma unroll
            for (int nf = 0; nf < 4; ++nf) {
                acc[mf][nf] = __builtin_amdgcn_wmma_f32_16x16x32_bf16(
                    false, ahi[mf].v, false, bhi[nf].v, (short)0, acc[mf][nf], false, false);
                acc[mf][nf] = __builtin_amdgcn_wmma_f32_16x16x32_bf16(
                    false, ahi[mf].v, false, blo[nf].v, (short)0, acc[mf][nf], false, false);
                acc[mf][nf] = __builtin_amdgcn_wmma_f32_16x16x32_bf16(
                    false, alo[mf].v, false, bhi[nf].v, (short)0, acc[mf][nf], false, false);
            }
        }
    };

    // k-loop unrolled x2 so the stage offset is a literal at each call site
    for (int kt = 0; kt < K_TILES; kt += 2) {
        step(kt,     0);
        step(kt + 1, 2048);
    }

    // ---- epilogue: C/D 16x16 f32 layout -> S[m, n] row-major ----
    const int col_lo = lane & 15;
    const int row_hi = (lane & 16) ? 8 : 0;
#pragma unroll
    for (int mf = 0; mf < 2; ++mf) {
#pragma unroll
        for (int nf = 0; nf < 4; ++nf) {
            int col = bn * 128 + wn * 64 + nf * 16 + col_lo;
#pragma unroll
            for (int v = 0; v < 8; ++v) {
                int row = m_base + wm * 32 + mf * 16 + v + row_hi;
                S[(size_t)row * N_TOT + col] = acc[mf][nf][v];
            }
        }
    }
}

// ---------------------------------------------------------------------------
// Kernel 3: overlap-add + window-sum normalize + slice [1024 : 1024+LEN_OUT)
// out[b, j] = ( sum_t S[b, t, n - 512t] ) / wsum[n],  n = j + 1024
// ---------------------------------------------------------------------------
__global__ __launch_bounds__(256) void istft_ola_kernel(
    const float* __restrict__ S, const float* __restrict__ wsum,
    float* __restrict__ out)
{
    long idx = (long)blockIdx.x * 256 + threadIdx.x;
    if (idx >= (long)B_SZ * LEN_OUT) return;
    int b = (int)(idx / LEN_OUT);
    int j = (int)(idx - (long)b * LEN_OUT);
    int n = j + (NFFT / 2);

    int t1 = n >> 9;                 // n / HOP
    if (t1 > T_SZ - 1) t1 = T_SZ - 1;
    int t0 = n - (NFFT - 1);
    t0 = (t0 < 0) ? 0 : ((t0 + HOP - 1) >> 9);

    float acc = 0.f;
    for (int t = t0; t <= t1; ++t)
        acc += S[((size_t)(b * T_SZ + t)) * N_TOT + (n - (t << 9))];

    out[idx] = acc / wsum[n];
}

// ---------------------------------------------------------------------------
extern "C" void kernel_launch(void* const* d_in, const int* in_sizes, int n_in,
                              void* d_out, int out_size, void* d_ws, size_t ws_size,
                              hipStream_t stream) {
    const float* re   = (const float*)d_in[0];  // real_stft (8,1,1024,1025)
    const float* im   = (const float*)d_in[1];  // imag_stft
    const float* Cr   = (const float*)d_in[2];  // (2048,2048)
    const float* Ci   = (const float*)d_in[3];  // (2048,2048)
    const float* wsum = (const float*)d_in[4];  // (525824)
    float* out = (float*)d_out;

    unsigned char* wsb = (unsigned char*)d_ws;
    unsigned* Whi = (unsigned*)(wsb);
    unsigned* Wlo = (unsigned*)(wsb + WLO_OFF);
    float*    S   = (float*)(wsb + S_OFF);

    // 1) fold + split weights into B-fragment layout
    prep_w_kernel<<<K_TILES * NT_TOT, 256, 0, stream>>>(Cr, Ci, Whi, Wlo);

    // 2) WMMA GEMM -> S[8192, 2048]
    dim3 grid(M_TOT / 128, N_TOT / 128);
    istft_gemm_kernel<<<grid, 256, 0, stream>>>(re, im, Whi, Wlo, S);

    // 3) overlap-add + normalize -> out[8, 523776]
    long total = (long)B_SZ * LEN_OUT;
    int blocks = (int)((total + 255) / 256);
    istft_ola_kernel<<<blocks, 256, 0, stream>>>(S, wsum, out);
}
